// Attention_FRN_26310969655745
// MI455X (gfx1250) — matrix operand order repaired and checked by reading, not compiled, not run on hardware
//
#include <hip/hip_runtime.h>

// CDNA5 / gfx1250, wave32. WMMA bf16 16x16x32, f32 accumulate.
typedef __attribute__((ext_vector_type(16))) __bf16 v16bf;
typedef __attribute__((ext_vector_type(8)))  float  v8f;

// 16-bit A-matrix 16x32 element->K mapping (ISA 7.12.2):
// lane half (lane>=16) and per-lane element e (0..15), vgpr j=e/2, hi/lo p=e&1
__device__ __forceinline__ int kA_idx(int e, int half) {
    int j = e >> 1, p = e & 1;
    return ((j >> 2) << 4) + half * 8 + ((j & 3) << 1) + p;
}

// -------- Kernel 1: M[ci][cj] = sum_c Wq[ci,c] * Wk[cj,c]  (256x256x256) ----
// Wq = W_qkv1[:, 0:256], Wk = W_qkv2[:, 256:512]; row stride of both = 768.
// One wave per 16x16 output tile; K loop of 8 bf16 WMMAs.
__global__ __launch_bounds__(32)
void k1_weight_gram(const float* __restrict__ W1, const float* __restrict__ W2,
                    float* __restrict__ Mws) {
    const int lane = threadIdx.x & 31;
    const int half = lane >> 4;
    const int n16  = lane & 15;
    const int ci0  = blockIdx.x * 16;
    const int cj0  = blockIdx.y * 16;
    v8f acc = {};
    for (int kb = 0; kb < 256; kb += 32) {
        v16bf a, b;
#pragma unroll
        for (int e = 0; e < 16; ++e) {
            const int ka = kb + kA_idx(e, half);          // A[m=ci][k=c]
            a[e] = (__bf16)W1[(ci0 + n16) * 768 + ka];
            const int kk = kb + kA_idx(e, half);          // B[k=c][n=cj]
            b[e] = (__bf16)W2[(cj0 + n16) * 768 + 256 + kk];
        }
        acc = __builtin_amdgcn_wmma_f32_16x16x32_bf16(false, a, false, b,
                                                      (short)0, acc, false, false);
    }
#pragma unroll
    for (int i = 0; i < 8; ++i)
        Mws[(ci0 + half * 8 + i) * 256 + cj0 + n16] = acc[i];
}

// -------- Kernel 2: T[b,l,c] = sum_k HSI[b,k,l] * M[k,c]  (16384x256x256) ---
// A[l][k] = HSI[b][k][l] (lanes 0-15 read 16 consecutive l -> 64B coalesced).
// 8 waves / block, one 16x16 tile per wave, 2048 blocks = 16384 tiles.
__global__ __launch_bounds__(256)
void k2_project(const float* __restrict__ HSI, const float* __restrict__ Mws,
                float* __restrict__ Tws) {
    const int lane = threadIdx.x & 31;
    const int half = lane >> 4;
    const int n16  = lane & 15;
    const int tile = blockIdx.x * 8 + (threadIdx.x >> 5);
    const int ct = tile & 15;            // c tile (0..15)
    const int lt = (tile >> 4) & 255;    // l tile (0..255)
    const int b  = tile >> 12;           // batch  (0..3)
    const int l0 = lt * 16;
    const int c0 = ct * 16;
    v8f acc = {};
    for (int kb = 0; kb < 256; kb += 32) {
        v16bf a, bb;
#pragma unroll
        for (int e = 0; e < 16; ++e) {
            const int ka = kb + kA_idx(e, half);
            a[e]  = (__bf16)HSI[(size_t)(b * 256 + ka) * 4096 + l0 + n16];
            const int kk = kb + kA_idx(e, half);
            bb[e] = (__bf16)Mws[kk * 256 + c0 + n16];
        }
        acc = __builtin_amdgcn_wmma_f32_16x16x32_bf16(false, a, false, bb,
                                                      (short)0, acc, false, false);
    }
#pragma unroll
    for (int i = 0; i < 8; ++i)
        Tws[(size_t)(b * 4096 + l0 + half * 8 + i) * 256 + c0 + n16] = acc[i];
}

// -------- Kernel 3: attn[b,l,n] = T[b,l,:] . MSI[b,:,9l+n]; argmax -> offsets
// Block = one (b,row): 64 l's -> 576 contiguous m columns. 288 threads each
// own two adjacent m (float2 load): fully coalesced 2304B per c iteration.
// T tile (64x256 f32 = 64KB) staged in LDS; attn scratch aliases it after use.
__global__ __launch_bounds__(288)
void k3_attn_argmax(const float* __restrict__ MSI, const float* __restrict__ Tws,
                    float* __restrict__ out) {
    __shared__ float ldsT[64 * 256];   // exactly 64 KB; reused for attn scratch
    const int tid = threadIdx.x;
    const int b   = blockIdx.x >> 6;
    const int row = blockIdx.x & 63;
    const int l0  = row * 64;

    const float* Tbase = Tws + (size_t)(b * 4096 + l0) * 256;
    for (int idx = tid; idx < 64 * 256; idx += 288)
        ldsT[idx] = Tbase[idx];
    __syncthreads();

    const int t0 = 2 * tid;            // m-offsets owned by this thread
    const int t1 = t0 + 1;
    const int la = t0 / 9;             // l_local of t0
    const int lb = t1 / 9;             // l_local of t1
    const size_t m0 = (size_t)l0 * 9;  // = 576*row, contiguous per (b,c)

    float acc0 = 0.f, acc1 = 0.f;
#pragma unroll 4
    for (int c = 0; c < 256; ++c) {
        const float* mrow = MSI + (size_t)(b * 256 + c) * 36864 + m0;
        const float2 v = *(const float2*)(mrow + t0);
        acc0 = fmaf(v.x, ldsT[la * 256 + c], acc0);
        acc1 = fmaf(v.y, ldsT[lb * 256 + c], acc1);
    }
    __syncthreads();                   // done reading T tile
    ldsT[t0] = acc0;                   // alias: attn[576] in first 2304B
    ldsT[t1] = acc1;
    __syncthreads();

    if (tid < 64) {                    // one thread per l: argmax over n=0..8
        float best = ldsT[tid * 9];
        int bn = 0;
#pragma unroll
        for (int n = 1; n < 9; ++n) {
            const float v = ldsT[tid * 9 + n];
            if (v > best) { best = v; bn = n; }   // strict > => first-max tie rule
        }
        const size_t o = ((size_t)blockIdx.x * 64 + tid) * 2;
        out[o]     = (float)(bn / 3 - 1);
        out[o + 1] = (float)(bn % 3 - 1);
    }
}

extern "C" void kernel_launch(void* const* d_in, const int* in_sizes, int n_in,
                              void* d_out, int out_size, void* d_ws, size_t ws_size,
                              hipStream_t stream) {
    // inputs: 0=x (unused), 1=y (unused), 2=HSI_Patch, 3=MSI_Patch2, 4=W_qkv1, 5=W_qkv2
    const float* HSI = (const float*)d_in[2];
    const float* MSI = (const float*)d_in[3];
    const float* W1  = (const float*)d_in[4];
    const float* W2  = (const float*)d_in[5];
    float* Mws = (float*)d_ws;                     // 256*256*4   = 256 KB
    float* Tws = (float*)((char*)d_ws + 262144);   // 4*4096*256*4 = 16 MB
    // total workspace needed: ~17 MB (<< typical ws_size)

    k1_weight_gram<<<dim3(16, 16), 32, 0, stream>>>(W1, W2, Mws);
    k2_project<<<2048, 256, 0, stream>>>(HSI, Mws, Tws);
    k3_attn_argmax<<<256, 288, 0, stream>>>(MSI, Tws, (float*)d_out);
}
